// GINConvNet_42855183679500
// MI455X (gfx1250) — compile-verified, matrix operand order checked by reading, not run on hardware
//
#include <hip/hip_runtime.h>
#include <stdint.h>

typedef unsigned short u16;
typedef unsigned int   u32;
typedef __attribute__((ext_vector_type(16))) __bf16 v16bf;
typedef __attribute__((ext_vector_type(8)))  float  v8f;

union Frag {
  v16bf v;
  u16   s[16];
  u32   d[8];
  uint4 q[2];
};

__device__ __forceinline__ u16 f2bf(float f) {
  u32 u = __builtin_bit_cast(u32, f);
  u32 r = u + 0x7fffu + ((u >> 16) & 1u);
  return (u16)(r >> 16);
}
__device__ __forceinline__ u32 pk2bf(float lo, float hi) {
  return (u32)f2bf(lo) | ((u32)f2bf(hi) << 16);
}
__device__ __forceinline__ float bf2f(u16 b) {
  u32 u = ((u32)b) << 16;
  return __builtin_bit_cast(float, u);
}
// Element->K mapping for 16-bit A/B fragments (CDNA5 ISA 7.12.2, 16-bit A 16x32):
// lane half h: e<8 -> K = 8h+e ; e>=8 -> K = 16+8h+(e-8)
__device__ __forceinline__ int fkmap(int h, int e) {
  return (e < 8) ? (8 * h + e) : (16 + 8 * h + (e - 8));
}
__device__ __forceinline__ v8f wmma_bf16(const Frag& a, const Frag& b, v8f c) {
  return __builtin_amdgcn_wmma_f32_16x16x32_bf16(false, a.v, false, b.v, (short)0, c,
                                                 false, false);
}

// ---------------------------------------------------------------- utilities
__global__ void k_zero(float* __restrict__ p, int n) {
  int i = blockIdx.x * blockDim.x + threadIdx.x;
  if (i < n) p[i] = 0.f;
}
__global__ void k_copy4(float4* __restrict__ dst, const float4* __restrict__ src, int n4) {
  int i = blockIdx.x * blockDim.x + threadIdx.x;
  if (i < n4) dst[i] = src[i];
}

// Pack f32 weight matrix [K,N] (or [N,K] if trans) into bf16 B-fragment order:
// dst[((ks*NT + nt)*32 + lane)*16 + e] = W[k = ks*32+fk(h,e)][n = nt*16 + lane%16]
__global__ void k_pack_w(u16* __restrict__ dst, const float* __restrict__ src,
                         int Kreal, int N, int KS, int NT, int trans) {
  int i = blockIdx.x * blockDim.x + threadIdx.x;
  int total = KS * NT * 512;
  if (i >= total) return;
  int e = i & 15, lane = (i >> 4) & 31, t2 = i >> 9;
  int nt = t2 % NT, ks = t2 / NT;
  int h = lane >> 4, n = nt * 16 + (lane & 15);
  int k = ks * 32 + fkmap(h, e);
  float v = 0.f;
  if (k < Kreal) v = trans ? src[(size_t)n * Kreal + k] : src[(size_t)k * N + n];
  dst[i] = f2bf(v);
}

// ---------------------------------------------------------------- GNN pieces
template <int C>
__global__ void k_scatter(float* __restrict__ agg, const float* __restrict__ h,
                          const int* __restrict__ src, const int* __restrict__ dst,
                          int E) {
  long long i = (long long)blockIdx.x * blockDim.x + threadIdx.x;
  if (i >= (long long)E * C) return;
  int e = (int)(i / C), c = (int)(i % C);
  unsafeAtomicAdd(&agg[(size_t)dst[e] * C + c], h[(size_t)src[e] * C + c]);
}

__global__ void k_pool(float* __restrict__ pooled, const float* __restrict__ h,
                       const int* __restrict__ batch, int Nn) {
  long long i = (long long)blockIdx.x * blockDim.x + threadIdx.x;
  if (i >= (long long)Nn * 32) return;
  int node = (int)(i >> 5), c = (int)(i & 31);
  unsafeAtomicAdd(&pooled[(size_t)batch[node] * 32 + c], h[i]);
}

__global__ void k_bn_stats(const float* __restrict__ z, float* __restrict__ stats, int M) {
  __shared__ float ssum[256], ssq[256];
  int c = threadIdx.x & 31, rg = threadIdx.x >> 5;
  float s = 0.f, q = 0.f;
  for (int r = blockIdx.x * 8 + rg; r < M; r += gridDim.x * 8) {
    float v = z[(size_t)r * 32 + c];
    s += v; q += v * v;
  }
  ssum[threadIdx.x] = s; ssq[threadIdx.x] = q;
  __syncthreads();
  if (rg == 0) {
    for (int j = 1; j < 8; ++j) { s += ssum[j * 32 + c]; q += ssq[j * 32 + c]; }
    unsafeAtomicAdd(&stats[c], s);
    unsafeAtomicAdd(&stats[32 + c], q);
  }
}

__global__ void k_bn_apply(float* __restrict__ hout, const float* __restrict__ z,
                           const float* __restrict__ stats, const float* __restrict__ gamma,
                           const float* __restrict__ beta, int M) {
  long long i = (long long)blockIdx.x * blockDim.x + threadIdx.x;
  if (i >= (long long)M * 32) return;
  int c = (int)(i & 31);
  float inv = 1.f / (float)M;
  float mu = stats[c] * inv;
  float var = stats[32 + c] * inv - mu * mu;
  hout[i] = gamma[c] * (z[i] - mu) * rsqrtf(var + 1e-5f) + beta[c];
}

// ---------------------------------------------------------------- generic WMMA GEMM
// Y[M,N] = act(A[M,K] @ Wpacked + bias).
// One wave computes one 16-row m-tile x NTW n-tiles: A fragment built once per
// k-step and reused for NTW WMMAs (register-level A reuse).
template <bool A_BF16, bool RELU, bool OUT_BF16, int NTW>
__global__ __launch_bounds__(128) void k_gemm(const void* __restrict__ A_,
                                              const u16* __restrict__ Wp,
                                              const float* __restrict__ bias,
                                              void* __restrict__ Y_,
                                              int M, int Kreal, int KS, int NT,
                                              int ldA, int ldY) {
  const int NG = NT / NTW;                      // n-tile groups
  int tiles = ((M + 15) >> 4) * NG;
  int wid = blockIdx.x * 4 + (threadIdx.x >> 5);
  if (wid >= tiles) return;
  int lane = threadIdx.x & 31, l16 = lane & 15, h = lane >> 4;
  int mt = wid / NG, nt0 = (wid % NG) * NTW;
  int mrow = mt * 16 + l16;
  int mclamp = mrow < M ? mrow : M - 1;
  const bool kfull = (KS * 32 == Kreal);
  v8f acc[NTW];
  #pragma unroll
  for (int j = 0; j < NTW; ++j) acc[j] = (v8f){0.f, 0.f, 0.f, 0.f, 0.f, 0.f, 0.f, 0.f};
  for (int ks = 0; ks < KS; ++ks) {
    Frag a;
    if (A_BF16) {
      const u16* base = (const u16*)A_ + (size_t)mclamp * ldA + ks * 32 + 8 * h;
      a.q[0] = *(const uint4*)base;
      a.q[1] = *(const uint4*)(base + 16);
    } else {
      const float* row = (const float*)A_ + (size_t)mclamp * ldA;
      if (kfull) {  // contiguous taps: 4x float4 per fragment
        const float* base = row + ks * 32 + 8 * h;
        float4 f0 = ((const float4*)base)[0];
        float4 f1 = ((const float4*)base)[1];
        float4 f2 = ((const float4*)(base + 16))[0];
        float4 f3 = ((const float4*)(base + 16))[1];
        a.d[0] = pk2bf(f0.x, f0.y); a.d[1] = pk2bf(f0.z, f0.w);
        a.d[2] = pk2bf(f1.x, f1.y); a.d[3] = pk2bf(f1.z, f1.w);
        a.d[4] = pk2bf(f2.x, f2.y); a.d[5] = pk2bf(f2.z, f2.w);
        a.d[6] = pk2bf(f3.x, f3.y); a.d[7] = pk2bf(f3.z, f3.w);
      } else {
        #pragma unroll
        for (int e = 0; e < 16; ++e) {
          int k = ks * 32 + fkmap(h, e);
          a.s[e] = (k < Kreal) ? f2bf(row[k]) : (u16)0;
        }
      }
    }
    #pragma unroll
    for (int j = 0; j < NTW; ++j) {
      Frag b;
      const u16* wp = Wp + ((size_t)(ks * NT + nt0 + j) * 32 + lane) * 16;
      b.q[0] = ((const uint4*)wp)[0];
      b.q[1] = ((const uint4*)wp)[1];
      acc[j] = wmma_bf16(a, b, acc[j]);
    }
  }
  const bool mfull = (mt * 16 + 15 < M);
  #pragma unroll
  for (int j = 0; j < NTW; ++j) {
    int n = (nt0 + j) * 16 + l16;
    float bi = bias[n];
    #pragma unroll
    for (int r = 0; r < 8; ++r) {
      int m = mt * 16 + r + 8 * h;
      if (mfull || m < M) {
        float v = acc[j][r] + bi;
        if (RELU) v = fmaxf(v, 0.f);
        if (OUT_BF16) ((u16*)Y_)[(size_t)m * ldY + n] = f2bf(v);
        else          ((float*)Y_)[(size_t)m * ldY + n] = v;
      }
    }
  }
}

template <bool AB, bool RELU, bool OB, int NTW>
static void gemm(hipStream_t s, const void* A, const u16* Wp, const float* bias, void* Y,
                 int M, int Kreal, int KS, int NT, int ldA, int ldY) {
  int tiles = ((M + 15) / 16) * (NT / NTW);
  k_gemm<AB, RELU, OB, NTW><<<(tiles + 3) / 4, 128, 0, s>>>(A, Wp, bias, Y, M, Kreal, KS, NT, ldA, ldY);
}

// ---------------------------------------------------------------- fused per-graph CNN
// One workgroup per graph. LDS plan (48 KB, overlaid; stages separated by barriers):
//   region1 [0,18432):    s_x (736 f32) + p1 (32x242 bf16, ch-major) -> later p2 (64x78 bf16, ch-major)
//   region2 [18432,49152): o2 (240x64 bf16, pos-major)               -> later o3 (80x128 bf16, pos-major)
// Pad rows over-read/over-write only inside the static LDS block; discarded downstream.
__global__ __launch_bounds__(128) void k_cnn(const float* __restrict__ xo,
                                             const float* __restrict__ w1,
                                             const float* __restrict__ b1,
                                             const u16* __restrict__ w2p,
                                             const float* __restrict__ b2,
                                             const u16* __restrict__ w3p,
                                             const float* __restrict__ b3,
                                             u16* __restrict__ xt_in) {
  __shared__ __align__(16) char smem[49152];
  float* s_x  = (float*)smem;            // 736 f32
  u16*   s_p1 = (u16*)(smem + 2944);     // 32*242  [c][q]
  u16*   s_p2 = (u16*)smem;              // 64*78   [c][q]
  u16*   s_o2 = (u16*)(smem + 18432);    // 240*64  [p][c]
  u16*   s_o3 = (u16*)(smem + 18432);    // 80*128  [p][c]
  const int g = blockIdx.x, tid = threadIdx.x;
  const int wave = tid >> 5, lane = tid & 31, l16 = lane & 15, h = lane >> 4;

  // A: stage xo
  for (int i = tid; i < 736; i += 128) s_x[i] = (i < 735) ? xo[(size_t)g * 735 + i] : 0.f;
  __syncthreads();

  // B: conv1 (1->32, k=8) + relu + maxpool3 -> p1[c][q], q<242
  for (int idx = tid; idx < 32 * 242; idx += 128) {
    int c = idx / 242, q = idx % 242;
    float m = 0.f;  // relu output >= 0
    #pragma unroll
    for (int j = 0; j < 3; ++j) {
      int p = 3 * q + j;
      float acc = b1[c];
      #pragma unroll
      for (int t = 0; t < 8; ++t) acc += s_x[p + t] * w1[c * 8 + t];
      m = fmaxf(m, fmaxf(acc, 0.f));
    }
    s_p1[c * 242 + q] = f2bf(m);
  }
  __syncthreads();

  // C: conv2 as im2col GEMM: M=235 (15 mtiles), K=256 (8 ks), N=64 (all 4 nt per wave).
  // k = ks*32 + fk(h,e): each 8-element half has fixed channel ci, taps t=0..7 contiguous.
  for (int mt = wave; mt < 15; mt += 4) {
    int m = mt * 16 + l16;
    v8f acc[4];
    #pragma unroll
    for (int j = 0; j < 4; ++j) acc[j] = (v8f){0.f, 0.f, 0.f, 0.f, 0.f, 0.f, 0.f, 0.f};
    for (int ks = 0; ks < 8; ++ks) {
      Frag a;
      const u16* r0 = s_p1 + (ks * 4 + h)     * 242 + m;
      const u16* r1 = s_p1 + (ks * 4 + 2 + h) * 242 + m;
      #pragma unroll
      for (int t = 0; t < 8; ++t) { a.s[t] = r0[t]; a.s[8 + t] = r1[t]; }
      #pragma unroll
      for (int j = 0; j < 4; ++j) {
        Frag b;
        const u16* wp = w2p + ((size_t)(ks * 4 + j) * 32 + lane) * 16;
        b.q[0] = ((const uint4*)wp)[0]; b.q[1] = ((const uint4*)wp)[1];
        acc[j] = wmma_bf16(a, b, acc[j]);
      }
    }
    #pragma unroll
    for (int j = 0; j < 4; ++j) {
      int n = j * 16 + l16;
      float bi = b2[n];
      #pragma unroll
      for (int r = 0; r < 8; ++r) {
        int p = mt * 16 + r + 8 * h;    // p <= 239 < 240 rows allocated
        s_o2[p * 64 + n] = f2bf(fmaxf(acc[j][r] + bi, 0.f));
      }
    }
  }
  __syncthreads();

  // D: maxpool3 over positions -> p2[c][q], q<78 (p1/s_x dead)
  for (int idx = tid; idx < 64 * 78; idx += 128) {
    int c = idx / 78, q = idx % 78;
    float m0 = fmaxf(bf2f(s_o2[(3 * q + 0) * 64 + c]),
               fmaxf(bf2f(s_o2[(3 * q + 1) * 64 + c]),
                     bf2f(s_o2[(3 * q + 2) * 64 + c])));
    s_p2[c * 78 + q] = f2bf(m0);
  }
  __syncthreads();

  // E: conv3 GEMM: M=71 (5 mtiles), K=512 (16 ks), N=128 (2 groups of 4 nt).
  for (int grp = wave; grp < 5 * 2; grp += 4) {
    int mt = grp >> 1, nt0 = (grp & 1) * 4;
    int m = mt * 16 + l16;
    v8f acc[4];
    #pragma unroll
    for (int j = 0; j < 4; ++j) acc[j] = (v8f){0.f, 0.f, 0.f, 0.f, 0.f, 0.f, 0.f, 0.f};
    for (int ks = 0; ks < 16; ++ks) {
      Frag a;
      const u16* r0 = s_p2 + (ks * 4 + h)     * 78 + m;
      const u16* r1 = s_p2 + (ks * 4 + 2 + h) * 78 + m;
      #pragma unroll
      for (int t = 0; t < 8; ++t) { a.s[t] = r0[t]; a.s[8 + t] = r1[t]; }
      #pragma unroll
      for (int j = 0; j < 4; ++j) {
        Frag b;
        const u16* wp = w3p + ((size_t)(ks * 8 + nt0 + j) * 32 + lane) * 16;
        b.q[0] = ((const uint4*)wp)[0]; b.q[1] = ((const uint4*)wp)[1];
        acc[j] = wmma_bf16(a, b, acc[j]);
      }
    }
    #pragma unroll
    for (int j = 0; j < 4; ++j) {
      int n = (nt0 + j) * 16 + l16;
      float bi = b3[n];
      #pragma unroll
      for (int r = 0; r < 8; ++r) {
        int p = mt * 16 + r + 8 * h;    // p <= 79 < 80 rows allocated
        s_o3[p * 128 + n] = f2bf(fmaxf(acc[j][r] + bi, 0.f));
      }
    }
  }
  __syncthreads();

  // F: maxpool3 + flatten (c*23+q) -> xt_in[g]
  for (int idx = tid; idx < 2944; idx += 128) {
    int c = idx / 23, q = idx % 23;
    float m0 = fmaxf(bf2f(s_o3[(3 * q + 0) * 128 + c]),
               fmaxf(bf2f(s_o3[(3 * q + 1) * 128 + c]),
                     bf2f(s_o3[(3 * q + 2) * 128 + c])));
    xt_in[(size_t)g * 2944 + idx] = f2bf(m0);
  }
}

// ---------------------------------------------------------------- output head (N=1)
__global__ void k_head(float* __restrict__ out, const u16* __restrict__ h2,
                       const float* __restrict__ w, const float* __restrict__ b, int B) {
  int g = blockIdx.x * blockDim.x + threadIdx.x;
  if (g >= B) return;
  const uint2* hp = (const uint2*)(h2 + (size_t)g * 128);
  float acc = b[0];
  #pragma unroll 8
  for (int j = 0; j < 32; ++j) {
    uint2 u = hp[j];
    acc += bf2f((u16)(u.x & 0xffffu)) * w[4 * j + 0];
    acc += bf2f((u16)(u.x >> 16))     * w[4 * j + 1];
    acc += bf2f((u16)(u.y & 0xffffu)) * w[4 * j + 2];
    acc += bf2f((u16)(u.y >> 16))     * w[4 * j + 3];
  }
  out[g] = acc;
}

// ---------------------------------------------------------------- host driver
extern "C" void kernel_launch(void* const* d_in, const int* in_sizes, int n_in,
                              void* d_out, int out_size, void* d_ws, size_t ws_size,
                              hipStream_t stream) {
  (void)n_in; (void)out_size; (void)ws_size;
  const float* x        = (const float*)d_in[0];
  const int*   ei       = (const int*)d_in[1];
  const int*   batch    = (const int*)d_in[2];
  const float* xo       = (const float*)d_in[3];
  const float* gin1_W1  = (const float*)d_in[4];
  const float* gin1_b1  = (const float*)d_in[5];
  const float* gin1_W2  = (const float*)d_in[6];
  const float* gin1_b2  = (const float*)d_in[7];
  const float* gins_W1  = (const float*)d_in[8];
  const float* gins_b1  = (const float*)d_in[9];
  const float* gins_W2  = (const float*)d_in[10];
  const float* gins_b2  = (const float*)d_in[11];
  const float* bn_gamma = (const float*)d_in[12];
  const float* bn_beta  = (const float*)d_in[13];
  const float* fc1xd_W  = (const float*)d_in[14];
  const float* fc1xd_b  = (const float*)d_in[15];
  const float* cnn1_W   = (const float*)d_in[16];
  const float* cnn1_b   = (const float*)d_in[17];
  const float* cnn2_W   = (const float*)d_in[18];
  const float* cnn2_b   = (const float*)d_in[19];
  const float* cnn3_W   = (const float*)d_in[20];
  const float* cnn3_b   = (const float*)d_in[21];
  const float* fc1xt_W  = (const float*)d_in[22];
  const float* fc1xt_b  = (const float*)d_in[23];
  const float* fc1_W    = (const float*)d_in[24];
  const float* fc1_b    = (const float*)d_in[25];
  const float* fc2_W    = (const float*)d_in[26];
  const float* fc2_b    = (const float*)d_in[27];
  const float* out_W    = (const float*)d_in[28];
  const float* out_b    = (const float*)d_in[29];

  const int Nn = in_sizes[0] / 78;
  const int E  = in_sizes[1] / 2;
  const int B  = in_sizes[3] / 735;
  const int* esrc = ei;
  const int* edst = ei + E;

  char* ws = (char*)d_ws;
  size_t cur = 0;
  auto alloc = [&](size_t bytes) { size_t o = cur; cur += (bytes + 255) & ~(size_t)255; return o; };

  u16* p_g1W1 = (u16*)(ws + alloc(3 * 2 * 512 * 2));
  u16* p_g1W2 = (u16*)(ws + alloc(1 * 2 * 512 * 2));
  u16* p_gW1  = (u16*)(ws + alloc(4 * 2 * 512 * 2));
  u16* p_gW2  = (u16*)(ws + alloc(4 * 2 * 512 * 2));
  u16* p_xd   = (u16*)(ws + alloc(1 * 8 * 512 * 2));
  u16* p_c2   = (u16*)(ws + alloc(8 * 4 * 512 * 2));
  u16* p_c3   = (u16*)(ws + alloc(16 * 8 * 512 * 2));
  u16* p_xt   = (u16*)(ws + alloc((size_t)92 * 8 * 512 * 2));
  u16* p_f1   = (u16*)(ws + alloc((size_t)8 * 64 * 512 * 2));
  u16* p_f2   = (u16*)(ws + alloc((size_t)32 * 8 * 512 * 2));
  float* bufA  = (float*)(ws + alloc((size_t)Nn * 78 * 4));  // agg78 / t scratch
  float* bufB  = (float*)(ws + alloc((size_t)Nn * 32 * 4));  // h
  float* bufC  = (float*)(ws + alloc((size_t)Nn * 32 * 4));  // agg / z
  float* stats = (float*)(ws + alloc(64 * 4));
  float* pooled = (float*)(ws + alloc((size_t)B * 32 * 4));
  u16* xc    = (u16*)(ws + alloc((size_t)B * 256 * 2));
  u16* xt_in = (u16*)(ws + alloc((size_t)B * 2944 * 2));
  u16* h1    = (u16*)(ws + alloc((size_t)B * 1024 * 2));
  u16* h2    = (u16*)(ws + alloc((size_t)B * 128 * 2));

  auto packs = [&](u16* dst, const float* src, int K, int N, int trans) {
    int KS = (K + 31) / 32, NT = N / 16;
    int total = KS * NT * 512;
    k_pack_w<<<(total + 255) / 256, 256, 0, stream>>>(dst, src, K, N, KS, NT, trans);
  };
  packs(p_g1W1, gin1_W1, 78, 32, 0);
  packs(p_g1W2, gin1_W2, 32, 32, 0);
  for (int i = 0; i < 4; ++i) {
    packs(p_gW1 + i * 1024, gins_W1 + i * 1024, 32, 32, 0);
    packs(p_gW2 + i * 1024, gins_W2 + i * 1024, 32, 32, 0);
  }
  packs(p_xd, fc1xd_W, 32, 128, 0);
  packs(p_c2, cnn2_W, 256, 64, 1);    // cnn2_W is [O=64][I*k=256] -> transposed source
  packs(p_c3, cnn3_W, 512, 128, 1);
  packs(p_xt, fc1xt_W, 2944, 128, 0);
  packs(p_f1, fc1_W, 256, 1024, 0);
  packs(p_f2, fc2_W, 1024, 128, 0);

  // ---- GIN layer 1 (K=78) ----
  {
    long long n = (long long)Nn * 78;
    k_copy4<<<(int)((n / 4 + 255) / 256), 256, 0, stream>>>((float4*)bufA, (const float4*)x, (int)(n / 4));
    long long tot = (long long)E * 78;
    k_scatter<78><<<(int)((tot + 255) / 256), 256, 0, stream>>>(bufA, x, esrc, edst, E);
  }
  gemm<false, true, false, 2>(stream, bufA, p_g1W1, gin1_b1, bufB, Nn, 78, 3, 2, 78, 32);
  gemm<false, true, false, 2>(stream, bufB, p_g1W2, gin1_b2, bufC, Nn, 32, 1, 2, 32, 32);
  k_zero<<<1, 64, 0, stream>>>(stats, 64);
  k_bn_stats<<<1024, 256, 0, stream>>>(bufC, stats, Nn);
  k_bn_apply<<<(Nn * 32 + 255) / 256, 256, 0, stream>>>(bufB, bufC, stats, bn_gamma, bn_beta, Nn);

  // ---- GIN layers 2..5 ----
  for (int L = 0; L < 4; ++L) {
    long long n = (long long)Nn * 32;
    k_copy4<<<(int)((n / 4 + 255) / 256), 256, 0, stream>>>((float4*)bufC, (const float4*)bufB, (int)(n / 4));
    long long tot = (long long)E * 32;
    k_scatter<32><<<(int)((tot + 255) / 256), 256, 0, stream>>>(bufC, bufB, esrc, edst, E);
    gemm<false, true, false, 2>(stream, bufC, p_gW1 + L * 1024, gins_b1 + L * 32, bufA, Nn, 32, 1, 2, 32, 32);
    gemm<false, true, false, 2>(stream, bufA, p_gW2 + L * 1024, gins_b2 + L * 32, bufC, Nn, 32, 1, 2, 32, 32);
    k_zero<<<1, 64, 0, stream>>>(stats, 64);
    k_bn_stats<<<1024, 256, 0, stream>>>(bufC, stats, Nn);
    k_bn_apply<<<(Nn * 32 + 255) / 256, 256, 0, stream>>>(bufB, bufC, stats,
                                                          bn_gamma + (L + 1) * 32,
                                                          bn_beta + (L + 1) * 32, Nn);
  }

  // ---- global_add_pool + drug head ----
  k_zero<<<(B * 32 + 255) / 256, 256, 0, stream>>>(pooled, B * 32);
  k_pool<<<(Nn * 32 + 255) / 256, 256, 0, stream>>>(pooled, bufB, batch, Nn);
  gemm<false, true, true, 4>(stream, pooled, p_xd, fc1xd_b, xc, B, 32, 1, 8, 32, 256);  // xc[:, :128]

  // ---- omics CNN branch (fused, LDS-resident) ----
  k_cnn<<<B, 128, 0, stream>>>(xo, cnn1_W, cnn1_b, p_c2, cnn2_b, p_c3, cnn3_b, xt_in);
  gemm<true, false, true, 4>(stream, xt_in, p_xt, fc1xt_b, xc + 128, B, 2944, 92, 8, 2944, 256); // xc[:,128:]

  // ---- combined head ----
  gemm<true, true, true, 4>(stream, xc, p_f1, fc1_b, h1, B, 256, 8, 64, 256, 1024);
  gemm<true, true, true, 4>(stream, h1, p_f2, fc2_b, h2, B, 1024, 32, 8, 1024, 128);
  k_head<<<(B + 255) / 256, 256, 0, stream>>>((float*)d_out, h2, out_W, out_b, B);
}
